// xLSTMBlock_14516989461037
// MI455X (gfx1250) — compile-verified, matrix operand order checked by reading, not run on hardware
//
#include <hip/hip_runtime.h>
#include <hip/hip_bf16.h>
#include <math.h>

// ---------------------------------------------------------------------------
// xLSTM block (2-layer sLSTM scan + proj/GELU/residual/LayerNorm) for gfx1250
//   - bf16 WMMA (v_wmma_f32_16x16x32_bf16) for all GEMMs, fp32 accumulate
//   - all weights kept bf16 (34 MB) -> L2 resident on MI455X (192 MB L2)
//   - persistent scan kernel, software-pipelined layers, 1 grid barrier/step
// ---------------------------------------------------------------------------

typedef __attribute__((ext_vector_type(16))) __bf16 v16bf;
typedef __attribute__((ext_vector_type(8)))  float  v8f;

#define NB 8
#define NT 1024
#define ND 1024
#define NH 1024
#define REC_WGS 64

__device__ __forceinline__ v8f wmma_bf16f32(v16bf a, v16bf b, v8f c) {
  // D = A(16x32) * B(32x16) + C, fp32 accumulate
  return __builtin_amdgcn_wmma_f32_16x16x32_bf16(false, a, false, b,
                                                 (short)0, c, false, false);
}

// -------------------------- fp32 -> bf16 convert ---------------------------
__global__ void cvt_f32_bf16(const float* __restrict__ s, __bf16* __restrict__ d, int n) {
  int i = blockIdx.x * blockDim.x + threadIdx.x;
  if (i < n) d[i] = (__bf16)s[i];
}

// ------------------------------ state init ---------------------------------
__global__ void init_state(__bf16* h0, __bf16* h1, unsigned* bar) {
  int i = blockIdx.x * blockDim.x + threadIdx.x;
  if (i < 2 * NB * NH) { h0[i] = (__bf16)0.0f; h1[i] = (__bf16)0.0f; }
  if (i == 0) *bar = 0u;
}

// --------------------------- persistent scan -------------------------------
// 64 WGs x 256 threads. WG b owns hidden slice j in [16b, 16b+16).
// Phase p: layer0 computes t=p (if p<NT), layer1 computes t=p-1 (if p>0).
// h0/h1 are double buffered [2][NB][NH] bf16. c-states live in registers.
__global__ __launch_bounds__(256, 1)
void slstm_scan(const __bf16* __restrict__ Xbf,
                const __bf16* __restrict__ Wih0, const __bf16* __restrict__ Whh0,
                const __bf16* __restrict__ Wih1, const __bf16* __restrict__ Whh1,
                const float*  __restrict__ b0,   const float*  __restrict__ b1,
                __bf16* __restrict__ h0buf, __bf16* __restrict__ h1buf,
                __bf16* __restrict__ lstmout, unsigned* __restrict__ bar)
{
  __shared__ float red[2 * 4 * 2 * 128];   // [layer][gate][khalf][m*16+j] (8 KB)

  const int tid  = threadIdx.x;
  const int wv   = tid >> 5;        // wave 0..7
  const int lane = tid & 31;
  const int j0   = blockIdx.x * 16; // hidden slice base
  const int gate = wv & 3;          // i/f/g/o
  const int kh   = wv >> 2;         // K half (0: k<512, 1: k>=512)
  const int arow = (lane & 15) & 7; // A row (batch, rows 8..15 duplicate 0..7)
  const int k16  = (lane >> 4) << 4;// lane-group K offset (0 or 16)
  const int bn   = lane & 15;       // B column within tile

  const int lhalf = tid >> 7;       // 0: layer0 elementwise, 1: layer1
  const int t2    = tid & 127;
  const int em    = t2 >> 4;        // batch 0..7
  const int ej    = t2 & 15;        // hidden within slice
  const float* biasp = lhalf ? b1 : b0;
  float cst = 0.0f;                 // cell state owned by this thread

  const int kbeg = kh * 512, kend = kh * 512 + 512;

#define KACC(Aptr, Bptr)                                  \
  for (int kb = kbeg; kb < kend; kb += 32) {              \
    v16bf a_ = *(const v16bf*)((Aptr) + kb);              \
    v16bf b_ = *(const v16bf*)((Bptr) + kb);              \
    acc = wmma_bf16f32(a_, b_, acc);                      \
  }

  for (int p = 0; p <= NT; ++p) {
    const int rb0 = (p + 1) & 1;   // holds h0(p-1)
    const int wb0 = p & 1;         // receives h0(p)
    const int rb1 = p & 1;         // holds h1(p-2)
    const int wb1 = (p + 1) & 1;   // receives h1(p-1)

    // ---- layer 0 GEMM: gates0 = x(p) @ Wih0^T + h0(p-1) @ Whh0^T ----
    if (p < NT) {
      v8f acc = {};
      const __bf16* A1 = Xbf  + ((size_t)arow * NT + p) * ND + k16;
      const __bf16* B1 = Wih0 + (size_t)(gate * NH + j0 + bn) * ND + k16;
      KACC(A1, B1)
      const __bf16* A2 = h0buf + (size_t)rb0 * (NB * NH) + (size_t)arow * NH + k16;
      const __bf16* B2 = Whh0 + (size_t)(gate * NH + j0 + bn) * NH + k16;
      KACC(A2, B2)
      if (lane < 16) {             // rows 0..7 valid (lanes 0-15, VGPRs 0-7)
        float* dst = &red[gate * 256 + kh * 128];
        #pragma unroll
        for (int r = 0; r < 8; ++r) dst[r * 16 + bn] = acc[r];
      }
    }
    // ---- layer 1 GEMM: gates1 = h0(p-1) @ Wih1^T + h1(p-2) @ Whh1^T ----
    if (p > 0) {
      v8f acc = {};
      const __bf16* A1 = h0buf + (size_t)rb0 * (NB * NH) + (size_t)arow * NH + k16;
      const __bf16* B1 = Wih1 + (size_t)(gate * NH + j0 + bn) * NH + k16;
      KACC(A1, B1)
      const __bf16* A2 = h1buf + (size_t)rb1 * (NB * NH) + (size_t)arow * NH + k16;
      const __bf16* B2 = Whh1 + (size_t)(gate * NH + j0 + bn) * NH + k16;
      KACC(A2, B2)
      if (lane < 16) {
        float* dst = &red[(4 + gate) * 256 + kh * 128];
        #pragma unroll
        for (int r = 0; r < 8; ++r) dst[r * 16 + bn] = acc[r];
      }
    }
    __syncthreads();

    // ---- elementwise sLSTM update (threads 0-127: layer0, 128-255: layer1)
    const bool doew = lhalf ? (p > 0) : (p < NT);
    if (doew) {
      const int lb = lhalf * 1024;
      const int colj = j0 + ej;
      const int mi = em * 16 + ej;
      float vi = red[lb +   0 + mi] + red[lb + 128 + mi] + biasp[0 * NH + colj];
      float vf = red[lb + 256 + mi] + red[lb + 384 + mi] + biasp[1 * NH + colj];
      float vg = red[lb + 512 + mi] + red[lb + 640 + mi] + biasp[2 * NH + colj];
      float vo = red[lb + 768 + mi] + red[lb + 896 + mi] + biasp[3 * NH + colj];
      float ig = expf(vi);                      // exponential input gate
      float fg = expf(vf);                      // exponential forget gate
      float gg = tanhf(vg);
      float og = 1.0f / (1.0f + expf(-vo));
      cst = fg * cst + ig * gg;
      float hv = og * tanhf(cst);
      __bf16 hb = (__bf16)hv;
      if (lhalf == 0) {
        h0buf[(size_t)wb0 * (NB * NH) + (size_t)em * NH + colj] = hb;
      } else {
        h1buf[(size_t)wb1 * (NB * NH) + (size_t)em * NH + colj] = hb;
        lstmout[((size_t)em * NT + (p - 1)) * NH + colj] = hb;
      }
    }

    // ---- grid-wide barrier (skip after last phase) ----
    if (p < NT) {
      __threadfence();
      __syncthreads();
      if (tid == 0) {
        __hip_atomic_fetch_add(bar, 1u, __ATOMIC_RELEASE, __HIP_MEMORY_SCOPE_AGENT);
        const unsigned target = (unsigned)(p + 1) * REC_WGS;
        while (__hip_atomic_load(bar, __ATOMIC_ACQUIRE, __HIP_MEMORY_SCOPE_AGENT) < target) {
          __builtin_amdgcn_s_sleep(8);
        }
      }
      __syncthreads();
    }
  }
#undef KACC
}

// ------------------- projection + GELU + residual + LN ---------------------
// 512 WGs x 256 threads; WG handles 16 rows of the flattened [8192,1024]
// lstm_out. Wave w owns columns [128w, 128w+128) via 8 WMMA N-tiles.
__global__ __launch_bounds__(256, 1)
void proj_gelu_ln(const __bf16* __restrict__ lstmout, const __bf16* __restrict__ PW,
                  const float* __restrict__ pb, const float* __restrict__ input,
                  const float* __restrict__ lng, const float* __restrict__ lnb,
                  float* __restrict__ out)
{
  __shared__ float xrow[16 * ND];   // 64 KB row buffer (gelu(proj)+residual)

  const int tid  = threadIdx.x;
  const int wv   = tid >> 5;
  const int lane = tid & 31;
  const int r0   = blockIdx.x * 16;     // flattened (b*T + t) row base
  const int n0   = wv * 128;
  const int arow = lane & 15;
  const int k16  = (lane >> 4) << 4;
  const int bn   = lane & 15;

  v8f acc[8] = {};
  for (int kb = 0; kb < NH; kb += 32) {
    v16bf a = *(const v16bf*)(lstmout + (size_t)(r0 + arow) * NH + kb + k16);
    #pragma unroll
    for (int t = 0; t < 8; ++t) {
      const int n = n0 + t * 16 + bn;
      v16bf b = *(const v16bf*)(PW + (size_t)n * NH + kb + k16);
      acc[t] = wmma_bf16f32(a, b, acc[t]);
    }
  }

  // epilogue: exact GELU + residual -> LDS
  const int mbase = (lane >> 4) * 8;   // lanes 16-31 hold rows 8..15
  #pragma unroll
  for (int t = 0; t < 8; ++t) {
    const int ncol = n0 + t * 16 + bn;
    const float pbv = pb[ncol];
    #pragma unroll
    for (int r = 0; r < 8; ++r) {
      const int m = mbase + r;
      float v  = acc[t][r] + pbv;
      float ge = 0.5f * v * (1.0f + erff(v * 0.70710678118654752f)); // exact GELU
      float x  = ge + input[(size_t)(r0 + m) * ND + ncol];           // residual
      xrow[m * ND + ncol] = x;
    }
  }
  __syncthreads();

  // LayerNorm: wave wv handles rows 2wv, 2wv+1
  for (int rr = wv * 2; rr < wv * 2 + 2; ++rr) {
    float s = 0.0f, s2 = 0.0f;
    for (int n = lane; n < ND; n += 32) {
      float x = xrow[rr * ND + n];
      s += x; s2 += x * x;
    }
    #pragma unroll
    for (int off = 16; off > 0; off >>= 1) {
      s  += __shfl_down(s, off, 32);
      s2 += __shfl_down(s2, off, 32);
    }
    s  = __shfl(s, 0, 32);
    s2 = __shfl(s2, 0, 32);
    const float mu  = s * (1.0f / ND);
    const float var = s2 * (1.0f / ND) - mu * mu;
    const float rs  = rsqrtf(var + 1e-5f);
    for (int n = lane; n < ND; n += 32) {
      float x = xrow[rr * ND + n];
      out[(size_t)(r0 + rr) * ND + n] = (x - mu) * rs * lng[n] + lnb[n];
    }
  }
}

// ------------------------------- launcher ----------------------------------
extern "C" void kernel_launch(void* const* d_in, const int* in_sizes, int n_in,
                              void* d_out, int out_size, void* d_ws, size_t ws_size,
                              hipStream_t stream) {
  const float* input = (const float*)d_in[0];
  const float* wih0  = (const float*)d_in[1];
  const float* whh0  = (const float*)d_in[2];
  const float* b0    = (const float*)d_in[3];
  const float* wih1  = (const float*)d_in[4];
  const float* whh1  = (const float*)d_in[5];
  const float* b1    = (const float*)d_in[6];
  const float* pw    = (const float*)d_in[7];
  const float* pb    = (const float*)d_in[8];
  const float* lng   = (const float*)d_in[9];
  const float* lnb   = (const float*)d_in[10];
  float* out = (float*)d_out;

  // workspace carve-up (~67 MB total)
  char* base = (char*)d_ws;
  size_t off = 0;
  auto carve = [&](size_t bytes) -> char* {
    char* p = base + off;
    off += (bytes + 255) & ~(size_t)255;
    return p;
  };
  const size_t W4 = (size_t)4 * NH * ND;        // 4H x 1024 elements
  __bf16* Wih0b = (__bf16*)carve(W4 * 2);
  __bf16* Whh0b = (__bf16*)carve(W4 * 2);
  __bf16* Wih1b = (__bf16*)carve(W4 * 2);
  __bf16* Whh1b = (__bf16*)carve(W4 * 2);
  __bf16* PWb   = (__bf16*)carve((size_t)ND * NH * 2);
  __bf16* Xbf   = (__bf16*)carve((size_t)NB * NT * ND * 2);
  __bf16* Lout  = (__bf16*)carve((size_t)NB * NT * NH * 2);
  __bf16* h0buf = (__bf16*)carve((size_t)2 * NB * NH * 2);
  __bf16* h1buf = (__bf16*)carve((size_t)2 * NB * NH * 2);
  unsigned* bar = (unsigned*)carve(256);
  (void)ws_size; (void)in_sizes; (void)n_in; (void)out_size;

  const int CT = 256;
  auto blocks = [](size_t n, int bs) { return (unsigned)((n + bs - 1) / bs); };

  // 1) convert weights + input to bf16 (recomputed each call: deterministic)
  cvt_f32_bf16<<<blocks(W4, CT), CT, 0, stream>>>(wih0, Wih0b, (int)W4);
  cvt_f32_bf16<<<blocks(W4, CT), CT, 0, stream>>>(whh0, Whh0b, (int)W4);
  cvt_f32_bf16<<<blocks(W4, CT), CT, 0, stream>>>(wih1, Wih1b, (int)W4);
  cvt_f32_bf16<<<blocks(W4, CT), CT, 0, stream>>>(whh1, Whh1b, (int)W4);
  cvt_f32_bf16<<<blocks((size_t)ND * NH, CT), CT, 0, stream>>>(pw, PWb, ND * NH);
  cvt_f32_bf16<<<blocks((size_t)NB * NT * ND, CT), CT, 0, stream>>>(input, Xbf, NB * NT * ND);

  // 2) zero h0/h1 double buffers + grid-barrier counter
  init_state<<<blocks(2 * NB * NH, CT), CT, 0, stream>>>(h0buf, h1buf, bar);

  // 3) persistent pipelined sLSTM scan (64 WGs, 1 grid barrier per step)
  slstm_scan<<<REC_WGS, CT, 0, stream>>>(Xbf, Wih0b, Whh0b, Wih1b, Whh1b,
                                         b0, b1, h0buf, h1buf, Lout, bar);

  // 4) projection + exact GELU + residual + LayerNorm
  proj_gelu_ln<<<(NB * NT) / 16, CT, 0, stream>>>(Lout, PWb, pb, input, lng, lnb, out);
}
